// EmergentCellularAutomaton_40553081209145
// MI455X (gfx1250) — matrix operand order repaired and checked by reading, not compiled, not run on hardware
//
#include <hip/hip_runtime.h>
#include <hip/hip_bf16.h>

// ---------------------------------------------------------------------------
// EmergentCellularAutomaton on MI455X (gfx1250, wave32, WMMA)
// B=4, N=4096, D=128, H=256, K-neighbors=8, 3 steps.
// Dominant cost: sim = keys@keys^T (~103 GFLOP f16-WMMA). simtopk v3:
// double-buffered GLOBAL_LOAD_ASYNC_TO_LDS_B128 B-tile staging (ASYNCcnt),
// LDS tile shared by 16 waves, fast-reject running-min top-k scan.
// ---------------------------------------------------------------------------

#define BATCH 4
#define NPTS  4096
#define DIN   128
#define HID   256
#define DC    384            // D + H
#define ROWS  (BATCH * NPTS) // 16384
#define KNBR  8
#define NSTEPS 3

#define STK_ROWS 128         // rows per simtopk block (8 row-tiles of 16)
#define STK_COLS 32          // columns per B-tile iteration

typedef __attribute__((ext_vector_type(16))) _Float16 v16h;
typedef __attribute__((ext_vector_type(8)))  _Float16 v8h;
typedef __attribute__((ext_vector_type(8)))  float    v8f;

// ---------------- WMMA fragment helpers (16x16x32 f16 -> f32) ---------------

__device__ __forceinline__ v8f wmma16(v16h a, v16h b, v8f c) {
  return __builtin_amdgcn_wmma_f32_16x16x32_f16(
      /*neg_a=*/false, a, /*neg_b=*/false, b,
      /*c_mod=*/(short)0, c, /*reuse_a=*/false, /*reuse_b=*/false);
}

// A tile: 16x32 (MxK) f16 row-major at base with row stride ld.
// Lane<16: row=lane, K 0..7 (elems 0..7) and K 16..23 (elems 8..15).
// Lane>=16: row=lane-16, K 8..15 and K 24..31.
__device__ __forceinline__ v16h load_a_frag(const _Float16* base, int ld) {
  int lane = threadIdx.x & 31;
  const _Float16* p = base + (size_t)(lane & 15) * ld + ((lane & 16) ? 8 : 0);
  v8h lo = *(const v8h*)p;
  v8h hi = *(const v8h*)(p + 16);
  return __builtin_shufflevector(lo, hi, 0,1,2,3,4,5,6,7,8,9,10,11,12,13,14,15);
}

// B tile given as B^T (NxK row-major): row n = output column, stride ld.
// Lane<16: col=lane, K 0..7 and K 16..23; lane>=16: col=lane-16, K 8..15 and K 24..31.
__device__ __forceinline__ v16h load_b_frag(const _Float16* baseT, int ld) {
  int lane = threadIdx.x & 31;
  const _Float16* p = baseT + (size_t)(lane & 15) * ld + ((lane & 16) ? 8 : 0);
  v8h lo = *(const v8h*)p;
  v8h hi = *(const v8h*)(p + 16);
  return __builtin_shufflevector(lo, hi, 0,1,2,3,4,5,6,7,8,9,10,11,12,13,14,15);
}

// C/D v8f mapping: element i -> row crow+i, col ccol
__device__ __forceinline__ void cd_coords(int& crow, int& ccol) {
  int lane = threadIdx.x & 31;
  crow = (lane & 16) ? 8 : 0;
  ccol = lane & 15;
}

// Async 32B copy global -> LDS (per lane), tracked by ASYNCcnt.
__device__ __forceinline__ void async_copy32(_Float16* lds_dst, const _Float16* gsrc) {
  unsigned lds = (unsigned)(uintptr_t)lds_dst;                 // DS offset = addr[31:0]
  unsigned long long ga = (unsigned long long)(uintptr_t)gsrc;
  asm volatile("global_load_async_to_lds_b128 %0, %1, off\n\t"
               "global_load_async_to_lds_b128 %0, %1, off offset:16"
               :: "v"(lds), "v"(ga) : "memory");
}

__device__ __forceinline__ void wait_async0() {
  asm volatile("s_wait_asynccnt 0x0" ::: "memory");
}

// ---------------- weight convert + transpose (f32 [r,c] -> f16 [c,r]) -------

__global__ void convtr_kernel(const float* __restrict__ W, _Float16* __restrict__ WT,
                              int rows, int cols) {
  int tid = blockIdx.x * blockDim.x + threadIdx.x;
  if (tid >= rows * cols) return;
  int r = tid / cols, c = tid % cols;
  WT[(size_t)c * rows + r] = (_Float16)W[tid];
}

// ---------------- state f32 -> f16 into combined[:, 0:128] (ld=384) ---------

__global__ void cast_state_kernel(const float* __restrict__ state,
                                  _Float16* __restrict__ combined) {
  int tid = blockIdx.x * 256 + threadIdx.x;       // ROWS*DIN elements
  int row = tid >> 7, col = tid & 127;
  combined[(size_t)row * DC + col] = (_Float16)state[tid];
}

// ---------------- keys = normalize(state @ Wn + bn) -> f16 ------------------
// Block: 16 rows x 256 cols, 8 waves, K=128.

__global__ __launch_bounds__(256)
void keys_kernel(const _Float16* __restrict__ combined,   // A (ld=DC), cols 0..127
                 const _Float16* __restrict__ WnT,        // 256x128 f16
                 const float* __restrict__ bn,
                 _Float16* __restrict__ keys_h) {         // ROWSxHID
  __shared__ float sk[16][HID + 8];
  __shared__ float rnorm[16];
  int m0 = blockIdx.x * 16;
  int wave = threadIdx.x >> 5;
  int crow, ccol; cd_coords(crow, ccol);
  for (int t = 0; t < 2; ++t) {
    int n0 = wave * 32 + t * 16;
    v8f acc = {};
#pragma unroll
    for (int k = 0; k < DIN; k += 32) {
      v16h a = load_a_frag(combined + (size_t)m0 * DC + k, DC);
      v16h b = load_b_frag(WnT + (size_t)n0 * DIN + k, DIN);
      acc = wmma16(a, b, acc);
    }
#pragma unroll
    for (int i = 0; i < 8; ++i)
      sk[crow + i][n0 + ccol] = acc[i] + bn[n0 + ccol];
  }
  __syncthreads();
  if (threadIdx.x < 16) {
    float s = 0.f;
    for (int j = 0; j < HID; ++j) { float v = sk[threadIdx.x][j]; s += v * v; }
    rnorm[threadIdx.x] = 1.f / fmaxf(sqrtf(s), 1e-12f);
  }
  __syncthreads();
  int r = threadIdx.x >> 4, c0 = (threadIdx.x & 15) * 16;
  float rn = rnorm[r];
#pragma unroll
  for (int j = 0; j < 16; ++j)
    keys_h[(size_t)(m0 + r) * HID + c0 + j] = (_Float16)(sk[r][c0 + j] * rn);
}

// ---------------- sim = keys @ keys^T, running top-8 per row ----------------
// Grid: (NPTS/128, BATCH). Block 512 thr = 16 waves covering 128 rows x 32
// cols per iteration. Double-buffered async B-tile staging in LDS: the DMA of
// tile i+1 overlaps the WMMA + top-k scan of tile i.

__global__ __launch_bounds__(512)
void simtopk_kernel(const _Float16* __restrict__ keys_h,
                    int* __restrict__ idx_out) {
  int b  = blockIdx.y;
  int m0 = blockIdx.x * STK_ROWS;                 // row base within batch
  const _Float16* Kb = keys_h + (size_t)b * NPTS * HID;

  __shared__ _Float16 Blds[2][STK_COLS][HID];       // 2 x 16 KB
  __shared__ float    ssc[STK_ROWS][STK_COLS + 1];  // ~17 KB
  __shared__ float    cval[STK_ROWS][32];           // 16 KB
  __shared__ int      cidx[STK_ROWS][32];           // 16 KB

  int wave = threadIdx.x >> 5;
  int wrow = (wave & 7) * 16;                     // row tile within block
  int wcol = (wave >> 3) * 16;                    // 0 or 16 within B tile
  int crow, ccol; cd_coords(crow, ccol);

  // Preload this wave's A fragments (16 rows x K=256): 8 x v16h in registers
  v16h afr[8];
#pragma unroll
  for (int k = 0; k < 8; ++k)
    afr[k] = load_a_frag(Kb + (size_t)(m0 + wrow) * HID + k * 32, HID);

  // per-thread staging slice of the B tile: 32 B
  int lc  = threadIdx.x >> 4;                     // 0..31 column
  int lh0 = (threadIdx.x & 15) * 16;              // 0..240 K offset

  // scanner: 4 threads per row, each scans 8 of the 32 columns per iter
  int srow = threadIdx.x >> 2;                    // 0..127
  int sq   = threadIdx.x & 3;                     // 0..3
  float tv[KNBR]; int ti[KNBR];
#pragma unroll
  for (int j = 0; j < KNBR; ++j) { tv[j] = -1e30f; ti[j] = 0; }
  float cmin = -1e30f;

  // prologue: async-stage tile 0
  async_copy32(&Blds[0][lc][lh0], Kb + (size_t)lc * HID + lh0);
  wait_async0();
  __syncthreads();

  const int NITER = NPTS / STK_COLS;
  for (int iter = 0; iter < NITER; ++iter) {
    int cur = iter & 1;
    // issue async DMA of the next tile into the spare buffer (overlaps compute)
    if (iter + 1 < NITER) {
      const _Float16* src = Kb + (size_t)((iter + 1) * STK_COLS + lc) * HID + lh0;
      async_copy32(&Blds[1 - cur][lc][lh0], src);
    }

    v8f acc = {};
#pragma unroll
    for (int k = 0; k < 8; ++k) {
      v16h bfr = load_b_frag(&Blds[cur][wcol][k * 32], HID);
      acc = wmma16(afr[k], bfr, acc);
    }
#pragma unroll
    for (int i = 0; i < 8; ++i)
      ssc[wrow + crow + i][wcol + ccol] = acc[i];
    __syncthreads();

    // fast-reject top-8 scan (1 compare per candidate in the common case)
    int cbase = iter * STK_COLS + sq * 8;
#pragma unroll
    for (int j = 0; j < 8; ++j) {
      float v = ssc[srow][sq * 8 + j];
      if (v > cmin) {
        int mi = 0; float mv = tv[0];
#pragma unroll
        for (int q = 1; q < KNBR; ++q) if (tv[q] < mv) { mv = tv[q]; mi = q; }
        tv[mi] = v; ti[mi] = cbase + j;
        cmin = tv[0];
#pragma unroll
        for (int q = 1; q < KNBR; ++q) cmin = fminf(cmin, tv[q]);
      }
    }
    wait_async0();        // next tile landed in LDS
    __syncthreads();      // all waves done scanning + waiting
  }

  // merge 4 partial top-8 lists per row (32 candidates), 128 threads active
#pragma unroll
  for (int j = 0; j < KNBR; ++j) {
    cval[srow][sq * 8 + j] = tv[j];
    cidx[srow][sq * 8 + j] = ti[j];
  }
  __syncthreads();
  if (threadIdx.x < STK_ROWS) {
    int r = threadIdx.x;
    for (int s = 0; s < KNBR; ++s) {
      int bi = 0; float bv = cval[r][0];
      for (int q = 1; q < 32; ++q) if (cval[r][q] > bv) { bv = cval[r][q]; bi = q; }
      idx_out[((size_t)b * NPTS + m0 + r) * KNBR + s] = cidx[r][bi];
      cval[r][bi] = -1e31f;
    }
  }
}

// ---------------- gathered mean of top-8 neighbors -> f16 -------------------

__global__ void gather_kernel(const float* __restrict__ state,
                              const int* __restrict__ idx,
                              _Float16* __restrict__ nbr_h) {
  int tid = blockIdx.x * 256 + threadIdx.x;       // ROWS*DIN
  int row = tid >> 7, col = tid & 127;
  int b = row >> 12;
  const float* Sb = state + ((size_t)b << 12) * DIN;
  float s = 0.f;
#pragma unroll
  for (int j = 0; j < KNBR; ++j) {
    int n = idx[(size_t)row * KNBR + j];
    s += Sb[(size_t)n * DIN + col];
  }
  nbr_h[tid] = (_Float16)(s * (1.0f / KNBR));
}

// ---------------- generic WMMA GEMM: C=f16(A@B + bias) ----------------------
// Grid: (M/16, N/128). 8 waves, each one 16-col tile.

__global__ __launch_bounds__(256)
void gemm_bias_f16_kernel(const _Float16* __restrict__ A, int lda,
                          const _Float16* __restrict__ BT, int ldb,
                          const float* __restrict__ bias,
                          _Float16* __restrict__ C, int ldc, int K) {
  int m0 = blockIdx.x * 16;
  int n0 = blockIdx.y * 128 + (threadIdx.x >> 5) * 16;
  v8f acc = {};
  for (int k = 0; k < K; k += 32) {
    v16h a = load_a_frag(A + (size_t)m0 * lda + k, lda);
    v16h b = load_b_frag(BT + (size_t)n0 * ldb + k, ldb);
    acc = wmma16(a, b, acc);
  }
  int crow, ccol; cd_coords(crow, ccol);
#pragma unroll
  for (int i = 0; i < 8; ++i) {
    int r = m0 + crow + i, c = n0 + ccol;
    C[(size_t)r * ldc + c] = (_Float16)(acc[i] + bias[c]);
  }
}

// ---------------- mlp1: LN(combined @ W1 + b1) -> silu -> f16 ---------------
// Block: 16 rows x 256 cols, K=384.

__global__ __launch_bounds__(256)
void mlp1_kernel(const _Float16* __restrict__ combined,
                 const _Float16* __restrict__ W1T,     // 256x384 f16
                 const float* __restrict__ b1,
                 const float* __restrict__ ln_g,
                 const float* __restrict__ ln_b,
                 _Float16* __restrict__ sh) {          // ROWSxHID
  __shared__ float sk[16][HID + 8];
  __shared__ float smu[16], srs[16];
  int m0 = blockIdx.x * 16;
  int wave = threadIdx.x >> 5;
  int crow, ccol; cd_coords(crow, ccol);
  for (int t = 0; t < 2; ++t) {
    int n0 = wave * 32 + t * 16;
    v8f acc = {};
#pragma unroll
    for (int k = 0; k < DC; k += 32) {
      v16h a = load_a_frag(combined + (size_t)m0 * DC + k, DC);
      v16h b = load_b_frag(W1T + (size_t)n0 * DC + k, DC);
      acc = wmma16(a, b, acc);
    }
#pragma unroll
    for (int i = 0; i < 8; ++i)
      sk[crow + i][n0 + ccol] = acc[i] + b1[n0 + ccol];
  }
  __syncthreads();
  if (threadIdx.x < 16) {
    float s = 0.f, s2 = 0.f;
    for (int j = 0; j < HID; ++j) { float v = sk[threadIdx.x][j]; s += v; s2 += v * v; }
    float mu = s * (1.0f / HID);
    float var = s2 * (1.0f / HID) - mu * mu;
    smu[threadIdx.x] = mu;
    srs[threadIdx.x] = rsqrtf(var + 1e-5f);
  }
  __syncthreads();
  int r = threadIdx.x >> 4, c0 = (threadIdx.x & 15) * 16;
  float mu = smu[r], rs = srs[r];
#pragma unroll
  for (int j = 0; j < 16; ++j) {
    int c = c0 + j;
    float h = (sk[r][c] - mu) * rs * ln_g[c] + ln_b[c];
    float sv = h / (1.f + __expf(-h));            // silu
    sh[(size_t)(m0 + r) * HID + c] = (_Float16)sv;
  }
}

// ---------------- mlp2: state += sh @ W2 + b2 (f32 residual) ----------------

__global__ __launch_bounds__(256)
void mlp2_kernel(const _Float16* __restrict__ sh,      // A, lda=HID
                 const _Float16* __restrict__ W2T,     // 128x256 f16
                 const float* __restrict__ b2,
                 float* __restrict__ state) {
  int m0 = blockIdx.x * 16;
  int n0 = (threadIdx.x >> 5) * 16;
  v8f acc = {};
#pragma unroll
  for (int k = 0; k < HID; k += 32) {
    v16h a = load_a_frag(sh + (size_t)m0 * HID + k, HID);
    v16h b = load_b_frag(W2T + (size_t)n0 * HID + k, HID);
    acc = wmma16(a, b, acc);
  }
  int crow, ccol; cd_coords(crow, ccol);
#pragma unroll
  for (int i = 0; i < 8; ++i) {
    int r = m0 + crow + i, c = n0 + ccol;
    state[(size_t)r * DIN + c] += acc[i] + b2[c];
  }
}

// ---------------- final: two-stage mean over N, then @ Wo + bo --------------

__global__ void mean1_kernel(const float* __restrict__ state,
                             float* __restrict__ partial) {
  int b = blockIdx.x, g = blockIdx.y, c = threadIdx.x;  // grid (4,32), 128 thr
  int r0 = g * 128;
  float s = 0.f;
  for (int n = 0; n < 128; ++n)
    s += state[((size_t)b * NPTS + r0 + n) * DIN + c];
  partial[((size_t)b * 32 + g) * DIN + c] = s;
}

__global__ void mean2_kernel(const float* __restrict__ partial,
                             float* __restrict__ mv) {
  int b = blockIdx.x, c = threadIdx.x;                  // grid 4, 128 thr
  float s = 0.f;
  for (int g = 0; g < 32; ++g)
    s += partial[((size_t)b * 32 + g) * DIN + c];
  mv[b * DIN + c] = s * (1.0f / NPTS);
}

__global__ void out_kernel(const float* __restrict__ mv,
                           const float* __restrict__ Wo,
                           const float* __restrict__ bo,
                           float* __restrict__ out) {
  int b = blockIdx.x, o = threadIdx.x;                  // grid 4, 128 thr
  float s = bo[o];
  for (int k = 0; k < DIN; ++k)
    s += mv[b * DIN + k] * Wo[k * 128 + o];
  out[b * 128 + o] = s;
}

// ---------------------------------------------------------------------------

extern "C" void kernel_launch(void* const* d_in, const int* in_sizes, int n_in,
                              void* d_out, int out_size, void* d_ws, size_t ws_size,
                              hipStream_t stream) {
  const float* x    = (const float*)d_in[0];
  const float* Wn   = (const float*)d_in[1];
  const float* bn   = (const float*)d_in[2];
  const float* W1   = (const float*)d_in[3];
  const float* b1   = (const float*)d_in[4];
  const float* ln_g = (const float*)d_in[5];
  const float* ln_b = (const float*)d_in[6];
  const float* W2   = (const float*)d_in[7];
  const float* b2   = (const float*)d_in[8];
  const float* Wo   = (const float*)d_in[9];
  const float* bo   = (const float*)d_in[10];
  float* out = (float*)d_out;

  // workspace carve-up (256B aligned)
  char* ws = (char*)d_ws;
  size_t off = 0;
  auto carve = [&](size_t bytes) {
    void* p = ws + off;
    off += (bytes + 255) & ~(size_t)255;
    return p;
  };
  float*     state    = (float*)    carve((size_t)ROWS * DIN * 4);
  _Float16*  combined = (_Float16*) carve((size_t)ROWS * DC  * 2);
  _Float16*  keys_h   = (_Float16*) carve((size_t)ROWS * HID * 2);
  _Float16*  nbr_h    = (_Float16*) carve((size_t)ROWS * DIN * 2);
  _Float16*  sh       = (_Float16*) carve((size_t)ROWS * HID * 2);
  int*       idx      = (int*)      carve((size_t)ROWS * KNBR * 4);
  _Float16*  WnT      = (_Float16*) carve((size_t)DIN * HID * 2);   // 256x128
  _Float16*  W1T      = (_Float16*) carve((size_t)DC  * HID * 2);   // 256x384
  _Float16*  W2T      = (_Float16*) carve((size_t)HID * DIN * 2);   // 128x256
  float*     partial  = (float*)    carve((size_t)BATCH * 32 * DIN * 4);
  float*     mv       = (float*)    carve((size_t)BATCH * DIN * 4);

  // init state = x
  hipMemcpyAsync(state, x, (size_t)ROWS * DIN * 4, hipMemcpyDeviceToDevice, stream);

  // weight convert+transpose (f32 [K,N] -> f16 [N,K])
  convtr_kernel<<<(DIN * HID + 255) / 256, 256, 0, stream>>>(Wn, WnT, DIN, HID);
  convtr_kernel<<<(DC  * HID + 255) / 256, 256, 0, stream>>>(W1, W1T, DC, HID);
  convtr_kernel<<<(HID * DIN + 255) / 256, 256, 0, stream>>>(W2, W2T, HID, DIN);

  for (int step = 0; step < NSTEPS; ++step) {
    cast_state_kernel<<<ROWS * DIN / 256, 256, 0, stream>>>(state, combined);
    keys_kernel<<<ROWS / 16, 256, 0, stream>>>(combined, WnT, bn, keys_h);
    simtopk_kernel<<<dim3(NPTS / STK_ROWS, BATCH), 512, 0, stream>>>(keys_h, idx);
    gather_kernel<<<ROWS * DIN / 256, 256, 0, stream>>>(state, idx, nbr_h);
    // neighbor_agg = nbr @ Wn + bn  -> combined[:, 128:384]
    gemm_bias_f16_kernel<<<dim3(ROWS / 16, HID / 128), 256, 0, stream>>>(
        nbr_h, DIN, WnT, DIN, bn, combined + DIN, DC, DIN);
    mlp1_kernel<<<ROWS / 16, 256, 0, stream>>>(combined, W1T, b1, ln_g, ln_b, sh);
    mlp2_kernel<<<ROWS / 16, 256, 0, stream>>>(sh, W2T, b2, state);
  }

  mean1_kernel<<<dim3(BATCH, 32), DIN, 0, stream>>>(state, partial);
  mean2_kernel<<<BATCH, DIN, 0, stream>>>(partial, mv);
  out_kernel<<<BATCH, 128, 0, stream>>>(mv, Wo, bo, out);
  (void)in_sizes; (void)n_in; (void)out_size; (void)ws_size;
}
